// ProGAT_34024730919281
// MI455X (gfx1250) — compile-verified, matrix operand order checked by reading, not run on hardware
//
#include <hip/hip_runtime.h>

#define B_    8
#define S_    512
#define NB_   23
#define INIT_ 512
#define E_    256
#define M_    (B_ * S_)      // 4096
#define NEGV  (-9e8f)

// ---------------------------------------------------------------- WMMA types
typedef __bf16 bf16_t;
typedef bf16_t v16bf __attribute__((ext_vector_type(16)));
typedef float  v8f   __attribute__((ext_vector_type(8)));

struct __align__(16) U4 { unsigned x, y, z, w; };
union Frag { v16bf v; U4 q[2]; };

#define GPTR __attribute__((address_space(1)))
#define LPTR __attribute__((address_space(3)))

// pointee type of the async-copy builtin params (GCC-style vector, per diagnostic)
typedef int v4i_gcc __attribute__((__vector_size__(16)));

#if __has_builtin(__builtin_amdgcn_global_load_async_to_lds_b128)
#define HAVE_ASYNC_COPY 1
#else
#define HAVE_ASYNC_COPY 0
#endif

__device__ __forceinline__ void wait_async0() {
#if __has_builtin(__builtin_amdgcn_s_wait_asynccnt)
  __builtin_amdgcn_s_wait_asynccnt(0);
#else
  asm volatile("s_wait_asynccnt 0x0" ::: "memory");
#endif
}

__device__ __forceinline__ unsigned short f2bf(float f) {
  union { float f; unsigned u; } x; x.f = f;
  unsigned r = x.u + 0x7FFFu + ((x.u >> 16) & 1u);
  return (unsigned short)(r >> 16);
}
__device__ __forceinline__ float bf2f(unsigned short h) {
  union { unsigned u; float f; } x; x.u = ((unsigned)h) << 16; return x.f;
}
__device__ __forceinline__ float lrelu_(float x) { return x > 0.f ? x : 0.01f * x; }
__device__ __forceinline__ float elu_(float x)   { return x > 0.f ? x : (__expf(x) - 1.f); }
__device__ __forceinline__ float sig_(float x)   { return 1.f / (1.f + __expf(-x)); }

// Per-lane fragment loader following ISA 7.12.2 (16-bit A 16x32, mirrored B):
// lanes 0-15 hold K = k0+{0..7} (q0) and k0+16+{0..7} (q1); lanes 16-31 the +8 halves.
__device__ __forceinline__ void load_frag(const unsigned short* base, int ld,
                                          int row, int k0, Frag& f) {
  int lane = threadIdx.x & 31;
  const unsigned short* p =
      base + (size_t)(row + (lane & 15)) * ld + k0 + ((lane >> 4) << 3);
  f.q[0] = *(const U4*)p;
  f.q[1] = *(const U4*)(p + 16);
}

__device__ __forceinline__ v8f wmma_bf16(const Frag& a, const Frag& b, v8f c) {
  return __builtin_amdgcn_wmma_f32_16x16x32_bf16(false, a.v, false, b.v,
                                                 (short)0, c, false, false);
}

// ---------------------------------------------------------------- kernels
__global__ void f32_to_bf16(const float* __restrict__ x,
                            unsigned short* __restrict__ y, size_t n) {
  size_t i = (size_t)blockIdx.x * 256 + threadIdx.x;
  if (i < n) y[i] = f2bf(x[i]);
}

// C[M,N] = act(A_bf16[M,K] @ W_bf16[N,K]^T + bias[N]); act: 0 none, 1 lrelu.
// block = 256 thr (8 waves); wave w owns rows blockIdx.x*128 + w*16, cols blockIdx.y*64..+63.
// K must be a multiple of 64. Hand-rotated 2-stage pipeline: two disjoint fragment
// register sets alternate (no copies, no in-loop branch), so loads of stage B overlap
// the 4 WMMAs of stage A and the compiler emits partial s_wait_loadcnt.
__global__ __launch_bounds__(256) void gemm_bias_act(
    const unsigned short* __restrict__ A, const unsigned short* __restrict__ W,
    const float* __restrict__ bias, float* __restrict__ C,
    unsigned short* __restrict__ Cbf, int M, int N, int K, int act) {
  int wave = threadIdx.x >> 5;
  int lane = threadIdx.x & 31;
  int row  = blockIdx.x * 128 + wave * 16;
  int col0 = blockIdx.y * 64;
  int n = lane & 15, hi = lane >> 4;

  v8f acc[4];
#pragma unroll
  for (int t = 0; t < 4; ++t) {
    float bv = bias ? bias[col0 + t * 16 + n] : 0.f;
#pragma unroll
    for (int r = 0; r < 8; ++r) acc[t][r] = bv;
  }

  Frag aA, bA[4], aB, bB[4];
  load_frag(A, K, row, 0, aA);
#pragma unroll
  for (int t = 0; t < 4; ++t) load_frag(W, K, col0 + t * 16, 0, bA[t]);

  int k0 = 0;
  for (; k0 + 64 < K; k0 += 64) {
    // stage B loads (k0+32), then stage A WMMAs (k0)
    load_frag(A, K, row, k0 + 32, aB);
#pragma unroll
    for (int t = 0; t < 4; ++t) load_frag(W, K, col0 + t * 16, k0 + 32, bB[t]);
    __builtin_prefetch(A + (size_t)(row + (lane & 15)) * K + k0 + 64, 0, 1);
#pragma unroll
    for (int t = 0; t < 4; ++t) acc[t] = wmma_bf16(aA, bA[t], acc[t]);
    // stage A loads (k0+64), then stage B WMMAs (k0+32)
    load_frag(A, K, row, k0 + 64, aA);
#pragma unroll
    for (int t = 0; t < 4; ++t) load_frag(W, K, col0 + t * 16, k0 + 64, bA[t]);
#pragma unroll
    for (int t = 0; t < 4; ++t) acc[t] = wmma_bf16(aB, bB[t], acc[t]);
  }
  // epilogue: k0 and k0+32 remain (k0 == K-64)
  load_frag(A, K, row, k0 + 32, aB);
#pragma unroll
  for (int t = 0; t < 4; ++t) load_frag(W, K, col0 + t * 16, k0 + 32, bB[t]);
#pragma unroll
  for (int t = 0; t < 4; ++t) acc[t] = wmma_bf16(aA, bA[t], acc[t]);
#pragma unroll
  for (int t = 0; t < 4; ++t) acc[t] = wmma_bf16(aB, bB[t], acc[t]);

#pragma unroll
  for (int t = 0; t < 4; ++t) {
    int c = col0 + t * 16 + n;
#pragma unroll
    for (int r = 0; r < 8; ++r) {
      int m = row + hi * 8 + r;
      float v = acc[t][r];
      if (act == 1) v = lrelu_(v);
      C[(size_t)m * N + c] = v;
      if (Cbf) Cbf[(size_t)m * N + c] = f2bf(v);
    }
  }
}

// Round-0 fused attention: one block per node (256 thr / 8 waves).
// Async-gathers 23 neighbor amino rows to LDS, computes nfeat (32x256, K=512) with WMMA,
// scores+softmax+weighted-sum on chip; emits reduced vector (bf16) and sum of weights.
__global__ __launch_bounds__(256) void attn0_kernel(
    const unsigned short* __restrict__ amino_bf, const float* __restrict__ feat,
    const int* __restrict__ idx, const unsigned short* __restrict__ nbW,
    const float* __restrict__ nb_b, const float* __restrict__ alignW0,
    const float* __restrict__ alignB0, unsigned short* __restrict__ vsum_bf,
    float* __restrict__ sumw_out) {
  __shared__ __align__(16) unsigned short Anb[32 * INIT_];  // 32 KB
  __shared__ __align__(16) unsigned short NFb[32 * E_];     // 16 KB
  __shared__ float rbuf[256];
  __shared__ float wsh[32];
  __shared__ int   nidx[32];
  __shared__ float c1sh;

  int node = blockIdx.x, tid = threadIdx.x;
  if (tid < 32) nidx[tid] = (tid < NB_) ? idx[node * NB_ + tid] : -1;
  __syncthreads();
  int bb = node >> 9;  // node / S_

  // gather neighbor rows (zero-pad rows 23..31 / invalid)
  for (int q = tid; q < 32 * (INIT_ / 8); q += 256) {
    int r = q >> 6, c8 = q & 63;
    int ni = nidx[r];
    unsigned short* dst = Anb + r * INIT_ + c8 * 8;
    if (r < NB_ && ni >= 0) {
      const unsigned short* src =
          amino_bf + ((size_t)(bb * S_ + ni)) * INIT_ + c8 * 8;
#if HAVE_ASYNC_COPY
      __builtin_amdgcn_global_load_async_to_lds_b128(
          (GPTR v4i_gcc*)src, (LPTR v4i_gcc*)dst, 0, 0);
#else
      *(U4*)dst = *(const U4*)src;
#endif
    } else {
      U4 z; z.x = z.y = z.z = z.w = 0u;
      *(U4*)dst = z;
    }
  }
#if HAVE_ASYNC_COPY
  wait_async0();
#endif
  // c1 = dot(feat[node], alignW0[0:E])
  rbuf[tid] = feat[(size_t)node * E_ + tid] * alignW0[tid];
  __syncthreads();
  for (int s = 128; s > 0; s >>= 1) {
    if (tid < s) rbuf[tid] += rbuf[tid + s];
    __syncthreads();
  }
  if (tid == 0) c1sh = rbuf[0];
  __syncthreads();

  // NF(32x256) = lrelu(Anb(32x512) @ nbW^T + nb_b) via WMMA, 8 waves x 4 tiles,
  // hand-rotated 2-stage pipeline over K (512 = multiple of 64).
  int wave = tid >> 5, lane = tid & 31;
  int n = lane & 15, hi = lane >> 4;
  int ct0 = wave * 2;
  v8f acc[2][2];
#pragma unroll
  for (int mt = 0; mt < 2; ++mt)
#pragma unroll
    for (int j = 0; j < 2; ++j)
#pragma unroll
      for (int r = 0; r < 8; ++r) acc[mt][j][r] = 0.f;

  Frag a0A, a1A, b0A, b1A, a0B, a1B, b0B, b1B;
  load_frag(Anb, INIT_, 0,  0, a0A);
  load_frag(Anb, INIT_, 16, 0, a1A);
  load_frag(nbW, INIT_, ct0 * 16,       0, b0A);
  load_frag(nbW, INIT_, (ct0 + 1) * 16, 0, b1A);
  int k0 = 0;
  for (; k0 + 64 < INIT_; k0 += 64) {
    load_frag(Anb, INIT_, 0,  k0 + 32, a0B);
    load_frag(Anb, INIT_, 16, k0 + 32, a1B);
    load_frag(nbW, INIT_, ct0 * 16,       k0 + 32, b0B);
    load_frag(nbW, INIT_, (ct0 + 1) * 16, k0 + 32, b1B);
    acc[0][0] = wmma_bf16(a0A, b0A, acc[0][0]);
    acc[1][0] = wmma_bf16(a1A, b0A, acc[1][0]);
    acc[0][1] = wmma_bf16(a0A, b1A, acc[0][1]);
    acc[1][1] = wmma_bf16(a1A, b1A, acc[1][1]);
    load_frag(Anb, INIT_, 0,  k0 + 64, a0A);
    load_frag(Anb, INIT_, 16, k0 + 64, a1A);
    load_frag(nbW, INIT_, ct0 * 16,       k0 + 64, b0A);
    load_frag(nbW, INIT_, (ct0 + 1) * 16, k0 + 64, b1A);
    acc[0][0] = wmma_bf16(a0B, b0B, acc[0][0]);
    acc[1][0] = wmma_bf16(a1B, b0B, acc[1][0]);
    acc[0][1] = wmma_bf16(a0B, b1B, acc[0][1]);
    acc[1][1] = wmma_bf16(a1B, b1B, acc[1][1]);
  }
  // epilogue: k0 and k0+32 remain (k0 == INIT_-64)
  load_frag(Anb, INIT_, 0,  k0 + 32, a0B);
  load_frag(Anb, INIT_, 16, k0 + 32, a1B);
  load_frag(nbW, INIT_, ct0 * 16,       k0 + 32, b0B);
  load_frag(nbW, INIT_, (ct0 + 1) * 16, k0 + 32, b1B);
  acc[0][0] = wmma_bf16(a0A, b0A, acc[0][0]);
  acc[1][0] = wmma_bf16(a1A, b0A, acc[1][0]);
  acc[0][1] = wmma_bf16(a0A, b1A, acc[0][1]);
  acc[1][1] = wmma_bf16(a1A, b1A, acc[1][1]);
  acc[0][0] = wmma_bf16(a0B, b0B, acc[0][0]);
  acc[1][0] = wmma_bf16(a1B, b0B, acc[1][0]);
  acc[0][1] = wmma_bf16(a0B, b1B, acc[0][1]);
  acc[1][1] = wmma_bf16(a1B, b1B, acc[1][1]);

#pragma unroll
  for (int mt = 0; mt < 2; ++mt)
#pragma unroll
    for (int j = 0; j < 2; ++j) {
      int c = (ct0 + j) * 16 + n;
      float bv = nb_b[c];
#pragma unroll
      for (int r = 0; r < 8; ++r) {
        int m = mt * 16 + hi * 8 + r;
        NFb[m * E_ + c] = f2bf(lrelu_(acc[mt][j][r] + bv));
      }
    }
  __syncthreads();

  // scores
  if (tid < NB_) {
    float s;
    if (nidx[tid] >= 0) {
      s = c1sh + alignB0[0];
      const unsigned short* rowp = NFb + tid * E_;
      for (int e = 0; e < E_; ++e) s += bf2f(rowp[e]) * alignW0[E_ + e];
      s = lrelu_(s);
    } else s = NEGV;
    wsh[tid] = s;
  }
  __syncthreads();
  if (tid == 0) {
    float mx = -3.4e38f;
    for (int i2 = 0; i2 < NB_; ++i2) mx = fmaxf(mx, wsh[i2]);
    float den = 0.f;
    for (int i2 = 0; i2 < NB_; ++i2) { float e2 = __expf(wsh[i2] - mx); wsh[i2] = e2; den += e2; }
    float inv = 1.f / den, sw = 0.f;
    for (int i2 = 0; i2 < NB_; ++i2) {
      float wv = wsh[i2] * inv * (nidx[i2] >= 0 ? 1.f : 0.f);
      wsh[i2] = wv; sw += wv;
    }
    sumw_out[node] = sw;
  }
  __syncthreads();
  float v = 0.f;
  for (int i2 = 0; i2 < NB_; ++i2) v += wsh[i2] * bf2f(NFb[i2 * E_ + tid]);
  vsum_bf[(size_t)node * E_ + tid] = f2bf(v);
}

// c1[m]=dot(X[m],w2[0:E]); g[m]=dot(X[m],w2[E:2E]); one block per row.
__global__ __launch_bounds__(256) void dotpair_kernel(
    const float* __restrict__ X, const float* __restrict__ w2,
    float* __restrict__ c1, float* __restrict__ g) {
  __shared__ float rbuf[256];
  int m = blockIdx.x, tid = threadIdx.x;
  float x = X[(size_t)m * E_ + tid];
  rbuf[tid] = x * w2[tid]; __syncthreads();
  for (int s = 128; s > 0; s >>= 1) { if (tid < s) rbuf[tid] += rbuf[tid + s]; __syncthreads(); }
  if (tid == 0) c1[m] = rbuf[0];
  __syncthreads();
  rbuf[tid] = x * w2[E_ + tid]; __syncthreads();
  for (int s = 128; s > 0; s >>= 1) { if (tid < s) rbuf[tid] += rbuf[tid + s]; __syncthreads(); }
  if (tid == 0) g[m] = rbuf[0];
}

// Rounds 1-2 attention (no projection): scores from precomputed dots, softmax,
// weighted neighbor sum of act; emits reduced bf16 vector + sum of weights.
__global__ __launch_bounds__(256) void attn_light(
    const float* __restrict__ act, const int* __restrict__ idx,
    const float* __restrict__ c1, const float* __restrict__ g,
    const float* __restrict__ alignBd, unsigned short* __restrict__ vsum_bf,
    float* __restrict__ sumw_out) {
  __shared__ int nidx[NB_ + 1];
  __shared__ float wsh[NB_ + 1];
  int node = blockIdx.x, tid = threadIdx.x;
  if (tid < NB_) nidx[tid] = idx[node * NB_ + tid];
  __syncthreads();
  int bb = node >> 9;
  if (tid == 0) {
    float sc[NB_];
    float mx = -3.4e38f, c1v = c1[node], bv = alignBd[0];
    for (int i2 = 0; i2 < NB_; ++i2) {
      int ni = nidx[i2];
      float s = (ni >= 0) ? lrelu_(c1v + g[bb * S_ + ni] + bv) : NEGV;
      sc[i2] = s; mx = fmaxf(mx, s);
    }
    float den = 0.f;
    for (int i2 = 0; i2 < NB_; ++i2) { float e2 = __expf(sc[i2] - mx); sc[i2] = e2; den += e2; }
    float inv = 1.f / den, sw = 0.f;
    for (int i2 = 0; i2 < NB_; ++i2) {
      float wv = sc[i2] * inv * (nidx[i2] >= 0 ? 1.f : 0.f);
      wsh[i2] = wv; sw += wv;
    }
    sumw_out[node] = sw;
  }
  __syncthreads();
  float v = 0.f;
  for (int i2 = 0; i2 < NB_; ++i2) {
    int ni = nidx[i2];
    if (ni >= 0) v += wsh[i2] * act[((size_t)(bb * S_ + ni)) * E_ + tid];
  }
  vsum_bf[(size_t)node * E_ + tid] = f2bf(v);
}

// ctx = elu(ctxpre + sumw*bias); store bf16 (ctx only feeds the GI GEMM)
__global__ void ctx_finish(const float* __restrict__ ctxpre,
                           const float* __restrict__ sumw,
                           const float* __restrict__ bias,
                           unsigned short* __restrict__ ctxbf) {
  int i = blockIdx.x * 256 + threadIdx.x;
  int m = i >> 8, e = i & 255;
  float v = ctxpre[i] + sumw[m] * bias[e];
  ctxbf[i] = f2bf(elu_(v));
}

__global__ void gru_combine(const float* __restrict__ GI, const float* __restrict__ GH,
                            const float* __restrict__ hprev, float* __restrict__ hout,
                            unsigned short* __restrict__ hbf, float* __restrict__ actf,
                            float* __restrict__ actextra) {
  int i = blockIdx.x * 256 + threadIdx.x;
  int m = i >> 8, e = i & 255;
  size_t base = (size_t)m * 3 * E_;
  float ir = GI[base + e], iz = GI[base + E_ + e], inn = GI[base + 2 * E_ + e];
  float hr = GH[base + e], hz = GH[base + E_ + e], hn = GH[base + 2 * E_ + e];
  float r = sig_(ir + hr), z = sig_(iz + hz);
  float nn = tanhf(inn + r * hn);
  float h = (1.f - z) * nn + z * hprev[i];
  hout[i] = h; hbf[i] = f2bf(h);
  float a = h > 0.f ? h : 0.f;
  actf[i] = a;
  if (actextra) actextra[i] = a;
}

__global__ __launch_bounds__(256) void seqsum_kernel(const float* __restrict__ act,
                                                     const float* __restrict__ mask,
                                                     float* __restrict__ sf) {
  int b = blockIdx.x, e = threadIdx.x;
  float s = 0.f;
  for (int s2 = 0; s2 < S_; ++s2)
    s += act[((size_t)(b * S_ + s2)) * E_ + e] * mask[b * S_ + s2];
  sf[b * E_ + e] = s;
}

// Sequence head: scores/weights/ctx are loop-invariant (act_mol fixed), GRU runs T times.
__global__ __launch_bounds__(256) void seqfinal_kernel(
    const float* __restrict__ sf_in, const float* __restrict__ act,
    const float* __restrict__ mask, const float* __restrict__ gseq,
    const float* __restrict__ saW, const float* __restrict__ saB,
    const float* __restrict__ satW, const float* __restrict__ satB,
    const float* __restrict__ wih, const float* __restrict__ bih,
    const float* __restrict__ whh, const float* __restrict__ bhh,
    float* __restrict__ out, int T) {
  __shared__ float rbuf[256];
  __shared__ float wsc[S_];
  __shared__ float vsh[E_], ctx[E_], sfsh[E_];
  __shared__ float gi[3 * E_], gh[3 * E_];
  __shared__ float sumwsh;
  int b = blockIdx.x, tid = threadIdx.x;
  float sf = sf_in[b * E_ + tid];
  float am = sf > 0.f ? sf : 0.f;
  rbuf[tid] = am * saW[tid]; __syncthreads();
  for (int s = 128; s > 0; s >>= 1) { if (tid < s) rbuf[tid] += rbuf[tid + s]; __syncthreads(); }
  float am1 = rbuf[0]; __syncthreads();
  for (int s2 = tid; s2 < S_; s2 += 256) {
    float sc = lrelu_(am1 + gseq[b * S_ + s2] + saB[0]);
    if (mask[b * S_ + s2] == 0.f) sc += NEGV;
    wsc[s2] = sc;
  }
  __syncthreads();
  if (tid == 0) {
    float mx = -3.4e38f;
    for (int s2 = 0; s2 < S_; ++s2) mx = fmaxf(mx, wsc[s2]);
    float den = 0.f;
    for (int s2 = 0; s2 < S_; ++s2) { float e2 = __expf(wsc[s2] - mx); wsc[s2] = e2; den += e2; }
    float inv = 1.f / den, sw = 0.f;
    for (int s2 = 0; s2 < S_; ++s2) {
      float wv = wsc[s2] * inv * mask[b * S_ + s2];
      wsc[s2] = wv; sw += wv;
    }
    sumwsh = sw;
  }
  __syncthreads();
  float v = 0.f;
  for (int s2 = 0; s2 < S_; ++s2) v += wsc[s2] * act[((size_t)(b * S_ + s2)) * E_ + tid];
  vsh[tid] = v; sfsh[tid] = sf;
  __syncthreads();
  float c = 0.f;
  for (int k = 0; k < E_; ++k) c += vsh[k] * satW[(size_t)tid * E_ + k];
  c += sumwsh * satB[tid];
  ctx[tid] = elu_(c);
  __syncthreads();
  for (int t = 0; t < T; ++t) {
    for (int o = tid; o < 3 * E_; o += 256) {
      float a = bih[o], h2 = bhh[o];
      const float* wr = wih + (size_t)o * E_;
      const float* hr2 = whh + (size_t)o * E_;
      for (int k = 0; k < E_; ++k) { a += ctx[k] * wr[k]; h2 += sfsh[k] * hr2[k]; }
      gi[o] = a; gh[o] = h2;
    }
    __syncthreads();
    float r = sig_(gi[tid] + gh[tid]);
    float z = sig_(gi[tid + E_] + gh[tid + E_]);
    float nn = tanhf(gi[tid + 2 * E_] + r * gh[tid + 2 * E_]);
    float hnew = (1.f - z) * nn + z * sfsh[tid];
    __syncthreads();
    sfsh[tid] = hnew;
    __syncthreads();
  }
  out[b * E_ + tid] = sfsh[tid] > 0.f ? sfsh[tid] : 0.f;
}

// ---------------------------------------------------------------- launch
extern "C" void kernel_launch(void* const* d_in, const int* in_sizes, int n_in,
                              void* d_out, int out_size, void* d_ws, size_t ws_size,
                              hipStream_t stream) {
  (void)in_sizes; (void)n_in; (void)out_size; (void)ws_size;
  const float* amino = (const float*)d_in[0];
  const float* mask  = (const float*)d_in[1];
  const float* embW  = (const float*)d_in[2];
  const float* emb_b = (const float*)d_in[3];
  const float* nbW   = (const float*)d_in[4];
  const float* nb_b  = (const float*)d_in[5];
  const float* alignW = (const float*)d_in[6];   // (3,1,512)
  const float* alignB = (const float*)d_in[7];   // (3,1)
  const float* attW   = (const float*)d_in[8];   // (3,256,256)
  const float* attB   = (const float*)d_in[9];   // (3,256)
  const float* wih    = (const float*)d_in[10];  // (3,768,256)
  const float* whh    = (const float*)d_in[11];
  const float* bih    = (const float*)d_in[12];  // (3,768)
  const float* bhh    = (const float*)d_in[13];
  const float* saW    = (const float*)d_in[14];  // (1,512)
  const float* saB    = (const float*)d_in[15];
  const float* satW   = (const float*)d_in[16];  // (256,256)
  const float* satB   = (const float*)d_in[17];
  const float* swih   = (const float*)d_in[18];  // (768,256)
  const float* sbih   = (const float*)d_in[19];
  const float* swhh   = (const float*)d_in[20];
  const float* sbhh   = (const float*)d_in[21];
  const int*   idx    = (const int*)d_in[22];

  char* cur = (char*)d_ws;
  auto alloc = [&](size_t bytes) -> void* {
    void* p = (void*)cur; cur += (bytes + 255) & ~(size_t)255; return p;
  };
  unsigned short* amino_bf = (unsigned short*)alloc((size_t)M_ * INIT_ * 2);
  unsigned short* embW_bf  = (unsigned short*)alloc((size_t)E_ * INIT_ * 2);
  unsigned short* nbW_bf   = (unsigned short*)alloc((size_t)E_ * INIT_ * 2);
  unsigned short* attW_bf  = (unsigned short*)alloc((size_t)3 * E_ * E_ * 2);
  unsigned short* wih_bf   = (unsigned short*)alloc((size_t)3 * 3 * E_ * E_ * 2);
  unsigned short* whh_bf   = (unsigned short*)alloc((size_t)3 * 3 * E_ * E_ * 2);
  float* feat_f            = (float*)alloc((size_t)M_ * E_ * 4);
  unsigned short* feat_bf  = (unsigned short*)alloc((size_t)M_ * E_ * 2);
  float* h_f               = (float*)alloc((size_t)M_ * E_ * 4);
  unsigned short* h_bf     = (unsigned short*)alloc((size_t)M_ * E_ * 2);
  float* act_f             = (float*)alloc((size_t)M_ * E_ * 4);
  float* c1                = (float*)alloc((size_t)M_ * 4);
  float* g                 = (float*)alloc((size_t)M_ * 4);
  unsigned short* vsum_bf  = (unsigned short*)alloc((size_t)M_ * E_ * 2);
  float* sumw              = (float*)alloc((size_t)M_ * 4);
  float* ctxpre            = (float*)alloc((size_t)M_ * E_ * 4);
  unsigned short* ctx_bf   = (unsigned short*)alloc((size_t)M_ * E_ * 2);
  float* GI                = (float*)alloc((size_t)M_ * 3 * E_ * 4);
  float* GH                = (float*)alloc((size_t)M_ * 3 * E_ * 4);
  float* seqfeat           = (float*)alloc((size_t)B_ * E_ * 4);

  dim3 blk(256);
  auto cv = [&](const float* x, unsigned short* y, size_t n) {
    f32_to_bf16<<<dim3((unsigned)((n + 255) / 256)), blk, 0, stream>>>(x, y, n);
  };
  cv(amino, amino_bf, (size_t)M_ * INIT_);
  cv(embW,  embW_bf,  (size_t)E_ * INIT_);
  cv(nbW,   nbW_bf,   (size_t)E_ * INIT_);
  cv(attW,  attW_bf,  (size_t)3 * E_ * E_);
  cv(wih,   wih_bf,   (size_t)3 * 3 * E_ * E_);
  cv(whh,   whh_bf,   (size_t)3 * 3 * E_ * E_);

  // feat = lrelu(amino @ embW^T + emb_b)
  gemm_bias_act<<<dim3(M_ / 128, E_ / 64), blk, 0, stream>>>(
      amino_bf, embW_bf, emb_b, feat_f, feat_bf, M_, E_, INIT_, 1);

  // ---- round 0
  attn0_kernel<<<dim3(M_), blk, 0, stream>>>(amino_bf, feat_f, idx, nbW_bf, nb_b,
                                             alignW, alignB, vsum_bf, sumw);
  gemm_bias_act<<<dim3(M_ / 128, E_ / 64), blk, 0, stream>>>(
      vsum_bf, attW_bf, nullptr, ctxpre, nullptr, M_, E_, E_, 0);
  ctx_finish<<<dim3(M_ * E_ / 256), blk, 0, stream>>>(ctxpre, sumw, attB, ctx_bf);
  gemm_bias_act<<<dim3(M_ / 128, (3 * E_) / 64), blk, 0, stream>>>(
      ctx_bf, wih_bf, bih, GI, nullptr, M_, 3 * E_, E_, 0);
  gemm_bias_act<<<dim3(M_ / 128, (3 * E_) / 64), blk, 0, stream>>>(
      feat_bf, whh_bf, bhh, GH, nullptr, M_, 3 * E_, E_, 0);
  gru_combine<<<dim3(M_ * E_ / 256), blk, 0, stream>>>(GI, GH, feat_f, h_f, h_bf,
                                                       act_f, nullptr);

  // ---- rounds 1..2
  float* out_act = (float*)d_out + B_ * E_;  // d_out = [act_seq (B*E), act (B*S*E)]
  for (int d = 1; d < 3; ++d) {
    dotpair_kernel<<<dim3(M_), blk, 0, stream>>>(act_f, alignW + (size_t)d * 2 * E_, c1, g);
    attn_light<<<dim3(M_), blk, 0, stream>>>(act_f, idx, c1, g, alignB + d, vsum_bf, sumw);
    gemm_bias_act<<<dim3(M_ / 128, E_ / 64), blk, 0, stream>>>(
        vsum_bf, attW_bf + (size_t)d * E_ * E_, nullptr, ctxpre, nullptr, M_, E_, E_, 0);
    ctx_finish<<<dim3(M_ * E_ / 256), blk, 0, stream>>>(ctxpre, sumw,
                                                        attB + (size_t)d * E_, ctx_bf);
    gemm_bias_act<<<dim3(M_ / 128, (3 * E_) / 64), blk, 0, stream>>>(
        ctx_bf, wih_bf + (size_t)d * 3 * E_ * E_, bih + (size_t)d * 3 * E_,
        GI, nullptr, M_, 3 * E_, E_, 0);
    gemm_bias_act<<<dim3(M_ / 128, (3 * E_) / 64), blk, 0, stream>>>(
        h_bf, whh_bf + (size_t)d * 3 * E_ * E_, bhh + (size_t)d * 3 * E_,
        GH, nullptr, M_, 3 * E_, E_, 0);
    gru_combine<<<dim3(M_ * E_ / 256), blk, 0, stream>>>(
        GI, GH, h_f, h_f, h_bf, act_f, (d == 2) ? out_act : nullptr);
  }

  // ---- sequence head
  seqsum_kernel<<<dim3(B_), blk, 0, stream>>>(act_f, mask, seqfeat);
  dotpair_kernel<<<dim3(M_), blk, 0, stream>>>(act_f, saW, c1, g);  // g = gseq
  seqfinal_kernel<<<dim3(B_), blk, 0, stream>>>(seqfeat, act_f, mask, g, saW, saB,
                                                satW, satB, swih, sbih, swhh, sbhh,
                                                (float*)d_out, 2);
}
